// MeshGraphNetLayer_v2_38345468018698
// MI455X (gfx1250) — matrix operand order repaired and checked by reading, not compiled, not run on hardware
//
#include <hip/hip_runtime.h>
#include <hip/hip_bf16.h>
#include <stdint.h>

#define H      128
#define AST    136     // edge act LDS stride (bf16 elems), 16B aligned rows, bank-friendly
#define NAST   264     // node act LDS stride (256 cols + pad)
#define WST    136     // weight LDS stride for 128-wide input
#define W1ST   264     // node layer-1 weight LDS stride (256-wide input)
#define LN_EPS 1e-5f

typedef __bf16 bf16_t;
typedef __attribute__((ext_vector_type(16))) __bf16 v16bf;
typedef __attribute__((ext_vector_type(8)))  __bf16 v8bf;
typedef __attribute__((ext_vector_type(8)))  float  v8f;

struct MlpParams {
  const float *W0, *b0, *W1, *b1, *W2, *b2, *W3, *b3, *g, *beta;
};

extern __shared__ char smem[];

__device__ __forceinline__ bf16_t f2bf(float f) {
  union { float f; uint32_t u; } c; c.f = f;
  uint32_t r = (c.u + 0x7FFFu + ((c.u >> 16) & 1u)) >> 16;   // round-to-nearest-even
  union { uint16_t u; bf16_t b; } o; o.u = (uint16_t)r;
  return o.b;
}

__device__ __forceinline__ v8f zero8() {
  v8f z;
#pragma unroll
  for (int i = 0; i < 8; i++) z[i] = 0.f;
  return z;
}

// A fragment: 16x32 bf16, lanes 0-15 = rows M, VGPR0-3 hold K 0-7 (lo half) /
// 8-15 (hi half), VGPR4-7 hold K 16-23 / 24-31.  Two ds_load_b128 per lane.
__device__ __forceinline__ v16bf load_a(const bf16_t* act, int stride, int k0, int lane) {
  const int half = (lane >> 4) & 1;
  const int m    = lane & 15;
  const bf16_t* p = act + m * stride + k0 + half * 8;
  v8bf lo = *(const v8bf*)(p);
  v8bf hi = *(const v8bf*)(p + 16);
  v16bf r;
#pragma unroll
  for (int i = 0; i < 8; i++) { r[i] = lo[i]; r[i + 8] = hi[i]; }
  return r;
}

// B fragment: 32x16 bf16, B[k][n] = W[n][k].  Lane (n = n0 + lane&15) reads
// K like the A fragment: lanes 0-15 K=k0..k0+7 / k0+16..k0+23, lanes 16-31
// K=k0+8..k0+15 / k0+24..k0+31.  Two ds_load_b128 per lane; padded stride avoids conflicts.
__device__ __forceinline__ v16bf load_b(const bf16_t* w, int stride, int n0, int k0, int lane) {
  const int half = (lane >> 4) & 1;
  const int n    = n0 + (lane & 15);
  const bf16_t* p = w + n * stride + k0 + half * 8;
  v8bf lo = *(const v8bf*)(p);
  v8bf hi = *(const v8bf*)(p + 16);
  v16bf r;
#pragma unroll
  for (int i = 0; i < 8; i++) { r[i] = lo[i]; r[i + 8] = hi[i]; }
  return r;
}

__device__ __forceinline__ v8f wmma_bf16(v16bf a, v16bf b, v8f c) {
  // v_wmma_f32_16x16x32_bf16  (neg_a, A, neg_b, B, c_mod, C, reuse_a, reuse_b)
  return __builtin_amdgcn_wmma_f32_16x16x32_bf16(false, a, false, b, (short)0, c,
                                                 false, false);
}

// One wave computes D[16x128] = A[16xIN] @ W^T  using 8 N-tiles x IN/32 K-steps.
template <int IN>
__device__ __forceinline__ void gemm_tile(const bf16_t* act, int astride,
                                          const bf16_t* w, int wstride,
                                          int lane, v8f acc[8]) {
#pragma unroll
  for (int n = 0; n < 8; n++) acc[n] = zero8();
#pragma unroll
  for (int k = 0; k < IN; k += 32) {
    v16bf a = load_a(act, astride, k, lane);
#pragma unroll
    for (int n = 0; n < 8; n++) {
      v16bf b = load_b(w, wstride, n * 16, k, lane);
      acc[n] = wmma_bf16(a, b, acc[n]);
    }
  }
}

// bias (+ReLU) fused into bf16 write-back to the wave-private act tile.
// C/D layout: acc[n][r] = D[row = r + (lane>=16)*8][col = n*16 + lane&15]
__device__ __forceinline__ void bias_relu_store(bf16_t* act, int astride,
                                                const float* bias, int lane,
                                                const v8f acc[8], bool relu) {
  const int half = (lane >> 4) & 1;
  const int c0   = lane & 15;
#pragma unroll
  for (int n = 0; n < 8; n++) {
    float bb = bias[n * 16 + c0];
#pragma unroll
    for (int r = 0; r < 8; r++) {
      float v = acc[n][r] + bb;
      if (relu) v = fmaxf(v, 0.f);
      act[(r + half * 8) * astride + n * 16 + c0] = f2bf(v);
    }
  }
}

// Final layer: bias -> LayerNorm(row) -> gamma/beta -> + residual -> global store.
__device__ __forceinline__ void ln_residual_store(const v8f acc[8], const float* bias,
                                                  const float* g, const float* beta,
                                                  const float* __restrict__ resid,
                                                  float* out, int row0, int nRows,
                                                  int lane) {
  const int half = (lane >> 4) & 1;
  const int c0   = lane & 15;
  float h[8][8];  // [n][r]
#pragma unroll
  for (int n = 0; n < 8; n++) {
    float bb = bias[n * 16 + c0];
#pragma unroll
    for (int r = 0; r < 8; r++) h[n][r] = acc[n][r] + bb;
  }
#pragma unroll
  for (int r = 0; r < 8; r++) {
    float s = 0.f, q = 0.f;
#pragma unroll
    for (int n = 0; n < 8; n++) { float v = h[n][r]; s += v; q += v * v; }
    // reduce across the 16 lanes sharing this row (masks stay inside the group)
#pragma unroll
    for (int m = 1; m <= 8; m <<= 1) {
      s += __shfl_xor(s, m, 32);
      q += __shfl_xor(q, m, 32);
    }
    const float mu   = s * (1.f / 128.f);
    const float var  = q * (1.f / 128.f) - mu * mu;
    const float rinv = rsqrtf(var + LN_EPS);
    const int grow = row0 + r + half * 8;
    if (grow < nRows) {
#pragma unroll
      for (int n = 0; n < 8; n++) {
        const int col = n * 16 + c0;
        float v = g[col] * (h[n][r] - mu) * rinv + beta[col];
        out[(size_t)grow * H + col] = resid[(size_t)grow * H + col] + v;
      }
    }
  }
}

__device__ __forceinline__ void stage_w(bf16_t* dst, int dstride, const float* src,
                                        int rows, int cols, int tid, int nthr) {
  for (int i = tid; i < rows * cols; i += nthr)
    dst[(i / cols) * dstride + (i % cols)] = f2bf(src[i]);
}
__device__ __forceinline__ void stage_v(float* dst, const float* src, int n,
                                        int tid, int nthr) {
  for (int i = tid; i < n; i += nthr) dst[i] = src[i];
}

// ---------------------------------------------------------------- edge MLP ---
__global__ __launch_bounds__(256)
void edge_mlp_kernel(const float* __restrict__ ein, float* __restrict__ eout,
                     MlpParams p, int nRows) {
  const int tid  = threadIdx.x;
  const int lane = tid & 31;
  const int wave = tid >> 5;

  bf16_t* Wl   = (bf16_t*)smem;                               // 4 * 128*WST bf16
  float*  bias = (float*)(smem + 4 * 128 * WST * 2);          // 4 * 128
  float*  g    = bias + 4 * 128;
  float*  beta = g + 128;
  bf16_t* acts = (bf16_t*)(beta + 128);                       // 8 waves * 16*AST

  stage_w(Wl + 0 * 128 * WST, WST, p.W0, 128, 128, tid, 256);
  stage_w(Wl + 1 * 128 * WST, WST, p.W1, 128, 128, tid, 256);
  stage_w(Wl + 2 * 128 * WST, WST, p.W2, 128, 128, tid, 256);
  stage_w(Wl + 3 * 128 * WST, WST, p.W3, 128, 128, tid, 256);
  stage_v(bias + 0 * 128, p.b0, 128, tid, 256);
  stage_v(bias + 1 * 128, p.b1, 128, tid, 256);
  stage_v(bias + 2 * 128, p.b2, 128, tid, 256);
  stage_v(bias + 3 * 128, p.b3, 128, tid, 256);
  stage_v(g, p.g, 128, tid, 256);
  stage_v(beta, p.beta, 128, tid, 256);
  __syncthreads();

  bf16_t* act = acts + wave * 16 * AST;  // wave-private: no barriers below
  const int nTiles = (nRows + 127) >> 7;
  for (int tile = blockIdx.x; tile < nTiles; tile += gridDim.x) {
    const int row0 = tile * 128 + wave * 16;
    for (int i = lane; i < 16 * H; i += 32) {  // coalesced 128B per 32 lanes
      int r = i >> 7, c = i & (H - 1);
      int gr = row0 + r; if (gr > nRows - 1) gr = nRows - 1;
      act[r * AST + c] = f2bf(ein[(size_t)gr * H + c]);
    }
    v8f acc[8];
    gemm_tile<128>(act, AST, Wl + 0 * 128 * WST, WST, lane, acc);
    bias_relu_store(act, AST, bias + 0 * 128, lane, acc, true);
    gemm_tile<128>(act, AST, Wl + 1 * 128 * WST, WST, lane, acc);
    bias_relu_store(act, AST, bias + 1 * 128, lane, acc, true);
    gemm_tile<128>(act, AST, Wl + 2 * 128 * WST, WST, lane, acc);
    bias_relu_store(act, AST, bias + 2 * 128, lane, acc, true);
    gemm_tile<128>(act, AST, Wl + 3 * 128 * WST, WST, lane, acc);
    ln_residual_store(acc, bias + 3 * 128, g, beta, ein, eout, row0, nRows, lane);
  }
}

// ----------------------------------------------------------- scatter mean ----
__global__ __launch_bounds__(256)
void scatter_kernel(const float* __restrict__ e_new, const long long* __restrict__ col,
                    float* agg, float* cnt, int nE) {
  const int e    = blockIdx.x * 8 + (threadIdx.x >> 5);
  const int lane = threadIdx.x & 31;
  if (e >= nE) return;
  const long long c = col[e];
  const float* src = e_new + (size_t)e * H;
  float* dst = agg + (size_t)c * H;
#pragma unroll
  for (int j = 0; j < H / 32; j++) atomicAdd(dst + lane + 32 * j, src[lane + 32 * j]);
  if (lane == 0) atomicAdd(cnt + c, 1.f);
}

// ---------------------------------------------------------------- node MLP ---
__global__ __launch_bounds__(256)
void node_mlp_kernel(const float* __restrict__ x, const float* agg,
                     const float* cnt, float* xout, MlpParams p, int nRows) {
  const int tid  = threadIdx.x;
  const int lane = tid & 31;
  const int wave = tid >> 5;

  bf16_t* W1   = (bf16_t*)smem;                                // 128*W1ST
  bf16_t* Wl   = (bf16_t*)(smem + 128 * W1ST * 2);             // 3 * 128*WST
  float*  bias = (float*)(smem + 128 * W1ST * 2 + 3 * 128 * WST * 2);
  float*  g    = bias + 4 * 128;
  float*  beta = g + 128;
  bf16_t* acts = (bf16_t*)(beta + 128);                        // 8 * 16*NAST

  stage_w(W1, W1ST, p.W0, 128, 256, tid, 256);
  stage_w(Wl + 0 * 128 * WST, WST, p.W1, 128, 128, tid, 256);
  stage_w(Wl + 1 * 128 * WST, WST, p.W2, 128, 128, tid, 256);
  stage_w(Wl + 2 * 128 * WST, WST, p.W3, 128, 128, tid, 256);
  stage_v(bias + 0 * 128, p.b0, 128, tid, 256);
  stage_v(bias + 1 * 128, p.b1, 128, tid, 256);
  stage_v(bias + 2 * 128, p.b2, 128, tid, 256);
  stage_v(bias + 3 * 128, p.b3, 128, tid, 256);
  stage_v(g, p.g, 128, tid, 256);
  stage_v(beta, p.beta, 128, tid, 256);
  __syncthreads();

  bf16_t* act = acts + wave * 16 * NAST;
  const int nTiles = (nRows + 127) >> 7;
  for (int tile = blockIdx.x; tile < nTiles; tile += gridDim.x) {
    const int row0 = tile * 128 + wave * 16;
    for (int i = lane; i < 16 * 256; i += 32) {   // concat [x | agg/cnt] -> bf16
      int r = i >> 8, c = i & 255;
      int gr = row0 + r; if (gr > nRows - 1) gr = nRows - 1;
      float v;
      if (c < H) v = x[(size_t)gr * H + c];
      else       v = agg[(size_t)gr * H + (c - H)] / fmaxf(cnt[gr], 1.f);
      act[r * NAST + c] = f2bf(v);
    }
    v8f acc[8];
    gemm_tile<256>(act, NAST, W1, W1ST, lane, acc);
    bias_relu_store(act, NAST, bias + 0 * 128, lane, acc, true);
    gemm_tile<128>(act, NAST, Wl + 0 * 128 * WST, WST, lane, acc);
    bias_relu_store(act, NAST, bias + 1 * 128, lane, acc, true);
    gemm_tile<128>(act, NAST, Wl + 1 * 128 * WST, WST, lane, acc);
    bias_relu_store(act, NAST, bias + 2 * 128, lane, acc, true);
    gemm_tile<128>(act, NAST, Wl + 2 * 128 * WST, WST, lane, acc);
    ln_residual_store(acc, bias + 3 * 128, g, beta, x, xout, row0, nRows, lane);
  }
}

__global__ void zero_kernel(float* p, long n) {
  long i = (long)blockIdx.x * blockDim.x + threadIdx.x;
  const long stride = (long)gridDim.x * blockDim.x;
  for (; i < n; i += stride) p[i] = 0.f;
}

// ------------------------------------------------------------------- launch --
extern "C" void kernel_launch(void* const* d_in, const int* in_sizes, int n_in,
                              void* d_out, int out_size, void* d_ws, size_t ws_size,
                              hipStream_t stream) {
  if (n_in < 23) return;
  const float*     x    = (const float*)d_in[0];
  const long long* eidx = (const long long*)d_in[1];   // int64 [2, nE]
  const float*     ea   = (const float*)d_in[2];

  MlpParams ep;
  ep.W0 = (const float*)d_in[3];  ep.b0 = (const float*)d_in[4];
  ep.W1 = (const float*)d_in[5];  ep.b1 = (const float*)d_in[6];
  ep.W2 = (const float*)d_in[7];  ep.b2 = (const float*)d_in[8];
  ep.W3 = (const float*)d_in[9];  ep.b3 = (const float*)d_in[10];
  ep.g  = (const float*)d_in[11]; ep.beta = (const float*)d_in[12];

  MlpParams np;
  np.W0 = (const float*)d_in[13]; np.b0 = (const float*)d_in[14];
  np.W1 = (const float*)d_in[15]; np.b1 = (const float*)d_in[16];
  np.W2 = (const float*)d_in[17]; np.b2 = (const float*)d_in[18];
  np.W3 = (const float*)d_in[19]; np.b3 = (const float*)d_in[20];
  np.g  = (const float*)d_in[21]; np.beta = (const float*)d_in[22];

  const int nN = in_sizes[0] / H;   // 100000
  const int nE = in_sizes[2] / H;   // 500000

  float* xout = (float*)d_out;                 // [nN, H]  (also agg accumulator)
  float* eout = xout + (size_t)nN * H;         // [nE, H]
  float* cnt  = (float*)d_ws;                  // [nN] counts (400 KB)

  // 1) zero agg (reuses d_out x-region) and counts
  zero_kernel<<<1024, 256, 0, stream>>>(xout, (long)nN * H);
  zero_kernel<<<256, 256, 0, stream>>>(cnt, (long)nN);

  // 2) edge MLP (+residual) -> eout
  const size_t smem_edge = (size_t)4 * 128 * WST * 2   // bf16 weights
                         + (size_t)(4 * 128 + 256) * 4 // biases + gamma/beta
                         + (size_t)8 * 16 * AST * 2;   // per-wave act tiles
  edge_mlp_kernel<<<512, 256, smem_edge, stream>>>(ea, eout, ep, nE);

  // 3) scatter-mean accumulation into xout/cnt
  scatter_kernel<<<(nE + 7) / 8, 256, 0, stream>>>(eout, eidx + nE, xout, cnt, nE);

  // 4) node MLP (+residual) overwrites xout; each row's agg is consumed
  //    (staged to LDS) before that row is written -> safe aliasing
  const size_t smem_node = (size_t)128 * W1ST * 2
                         + (size_t)3 * 128 * WST * 2
                         + (size_t)(4 * 128 + 256) * 4
                         + (size_t)8 * 16 * NAST * 2;
  node_mlp_kernel<<<256, 256, smem_node, stream>>>(x, xout, cnt, xout, np, nN);
}
